// AdaptiveLMHead_89034672046373
// MI455X (gfx1250) — matrix operand order repaired and checked by reading, not compiled, not run on hardware
//
#include <hip/hip_runtime.h>
#include <hip/hip_bf16.h>

// ---------------------------------------------------------------------------
// Adaptive LM-head chunked cross-entropy for MI455X (gfx1250, wave32, WMMA).
// Weights are transposed to K-fastest once at bf16-conversion time so every
// hot-loop staging copy is a contiguous 16B async global->LDS transfer; the
// GEMMs run v_wmma_f32_16x16x32_bf16 with 2x2 register blocking per wave and
// the logsumexp + target gather are fused out of the accumulators (no logits
// ever hit memory).
// ---------------------------------------------------------------------------

typedef __bf16 bf16_t;
typedef __attribute__((ext_vector_type(16))) __bf16 v16bf;
typedef __attribute__((ext_vector_type(8)))  __bf16 v8bf;
typedef __attribute__((ext_vector_type(8)))  float  v8f;
typedef int async_v4i __attribute__((vector_size(16)));

#define THREADS 256          // 8 waves of 32
#define M_BLK   64           // rows per block   (2 wave-rows of 32)
#define N_CHUNK 128          // cols per chunk   (4 wave-cols of 32)
#define KS      128          // K slice staged in LDS
#define LDA     (KS + 8)     // padded LDS stride (136 elems = 68 banks)

union V16 { v16bf v; v8bf h[2]; };

__device__ __forceinline__ v8f wmma_bf16(v16bf a, v16bf b, v8f c) {
  return __builtin_amdgcn_wmma_f32_16x16x32_bf16(
      false, a, false, b, (short)0, c, false, false);
}

// 16-byte global -> LDS copy; async path (tracked by ASYNCcnt) when available.
__device__ __forceinline__ void copy16_g2l(const bf16_t* g, bf16_t* l) {
#if defined(__gfx1250__) && __has_builtin(__builtin_amdgcn_global_load_async_to_lds_b128)
  __builtin_amdgcn_global_load_async_to_lds_b128(
      (__attribute__((address_space(1))) async_v4i*)g,
      (__attribute__((address_space(3))) async_v4i*)l, 0, 0);
#else
  *reinterpret_cast<uint4*>(l) = *reinterpret_cast<const uint4*>(g);
#endif
}

__device__ __forceinline__ void wait_g2l() {
#if defined(__gfx1250__) && __has_builtin(__builtin_amdgcn_global_load_async_to_lds_b128)
#if __has_builtin(__builtin_amdgcn_s_wait_asynccnt)
  __builtin_amdgcn_s_wait_asynccnt(0);
#else
  asm volatile("s_wait_asynccnt 0x0" ::: "memory");
#endif
#endif
}

// ------------------------------- conversion --------------------------------
__global__ __launch_bounds__(THREADS)
void cvt_f32_bf16(const float* __restrict__ in, bf16_t* __restrict__ out, int n) {
  int i = blockIdx.x * blockDim.x + threadIdx.x;
  int stride = gridDim.x * blockDim.x;
  for (; i < n; i += stride) out[i] = (bf16_t)in[i];
}

// out[N][K] (bf16, K-fastest) = transpose(in[K][N] f32). LDS-tiled 64x64.
__global__ __launch_bounds__(THREADS)
void cvt_transpose(const float* __restrict__ in, bf16_t* __restrict__ out,
                   int Kdim, int Ndim) {
  __shared__ bf16_t T[64][72];
  const int k0 = blockIdx.y * 64;
  const int n0 = blockIdx.x * 64;
#pragma unroll
  for (int i = 0; i < 16; ++i) {
    int e = threadIdx.x + i * THREADS;
    int r = e >> 6, c = e & 63;                 // r: k, c: n
    int n = n0 + c;
    float v = (n < Ndim) ? in[(size_t)(k0 + r) * Ndim + n] : 0.f;
    T[c][r] = (bf16_t)v;
  }
  __syncthreads();
#pragma unroll
  for (int i = 0; i < 16; ++i) {
    int e = threadIdx.x + i * THREADS;
    int r = e >> 6, c = e & 63;                 // r: n, c: k
    int n = n0 + r;
    if (n < Ndim) out[(size_t)n * Kdim + k0 + c] = T[r][c];
  }
}

__global__ __launch_bounds__(THREADS)
void init_state(float* __restrict__ row_m, float* __restrict__ row_s,
                float* __restrict__ row_t, int n) {
  int i = blockIdx.x * blockDim.x + threadIdx.x;
  if (i < n) { row_m[i] = -1e30f; row_s[i] = 0.f; row_t[i] = 0.f; }
}

// ----------------------- shared GEMM inner machinery -----------------------
// Stage A slice [M_BLK][KS] and B slice [N_CHUNK][KS] (both K-fastest).
template<int K>
__device__ __forceinline__ void stage_A(const bf16_t* A, bf16_t* As,
                                        int rowbase, int ks, int t) {
#pragma unroll
  for (int i = 0; i < M_BLK * KS / 8 / THREADS; ++i) {   // 4 iters
    int e = t + i * THREADS;
    int r = e >> 4, c = (e & 15) * 8;                    // KS/8 == 16
    copy16_g2l(&A[(size_t)(rowbase + r) * K + ks + c], &As[r * LDA + c]);
  }
}

template<int K>
__device__ __forceinline__ void stage_B(const bf16_t* Wt, bf16_t* Bs,
                                        int n0, int N, int ks, int t) {
#pragma unroll
  for (int i = 0; i < N_CHUNK * KS / 8 / THREADS; ++i) { // 8 iters
    int e = t + i * THREADS;
    int r = e >> 4, c = (e & 15) * 8;
    int n = n0 + r;
    if (n < N) copy16_g2l(&Wt[(size_t)n * K + ks + c], &Bs[r * LDA + c]);
  }
}

// 2x2 tile WMMA over one staged K slice (fully unrolled: 16 wmma).
__device__ __forceinline__ void slice_mma(const bf16_t* As, const bf16_t* Bs,
                                          int wr, int wc, int lh, int ll,
                                          v8f& a00, v8f& a01, v8f& a10, v8f& a11) {
#pragma unroll
  for (int kk = 0; kk < KS; kk += 32) {
    V16 a0, a1, b0, b1;
    const bf16_t* ap0 = &As[(wr * 32 + ll) * LDA + kk + lh * 8];
    const bf16_t* ap1 = ap0 + 16 * LDA;
    a0.h[0] = *reinterpret_cast<const v8bf*>(ap0);
    a0.h[1] = *reinterpret_cast<const v8bf*>(ap0 + 16);
    a1.h[0] = *reinterpret_cast<const v8bf*>(ap1);
    a1.h[1] = *reinterpret_cast<const v8bf*>(ap1 + 16);
    const bf16_t* bp0 = &Bs[(wc * 32 + ll) * LDA + kk + lh * 8];
    const bf16_t* bp1 = bp0 + 16 * LDA;
    b0.h[0] = *reinterpret_cast<const v8bf*>(bp0);
    b0.h[1] = *reinterpret_cast<const v8bf*>(bp0 + 16);
    b1.h[0] = *reinterpret_cast<const v8bf*>(bp1);
    b1.h[1] = *reinterpret_cast<const v8bf*>(bp1 + 16);
    a00 = wmma_bf16(a0.v, b0.v, a00);
    a01 = wmma_bf16(a0.v, b1.v, a01);
    a10 = wmma_bf16(a1.v, b0.v, a10);
    a11 = wmma_bf16(a1.v, b1.v, a11);
  }
}

// ------------------------------ proj GEMM ----------------------------------
// P[8192][NP] = A[8192][1024] @ Wt[NP][1024]^T  (bf16 in/out, f32 acc).
template<int NP>
__global__ __launch_bounds__(THREADS)
void proj_gemm(const bf16_t* __restrict__ A, const bf16_t* __restrict__ Wt,
               bf16_t* __restrict__ P) {
  constexpr int K = 1024;
  __shared__ bf16_t As[M_BLK * LDA];
  __shared__ bf16_t Bs[N_CHUNK * LDA];

  const int t = threadIdx.x;
  const int rowbase = blockIdx.x * M_BLK;
  const int wave = t >> 5, lane = t & 31;
  const int wr = wave >> 2, wc = wave & 3;
  const int lh = lane >> 4, ll = lane & 15;

  for (int n0 = 0; n0 < NP; n0 += N_CHUNK) {
    v8f a00 = {}, a01 = {}, a10 = {}, a11 = {};
    for (int ks = 0; ks < K; ks += KS) {
      __syncthreads();
      stage_A<K>(A, As, rowbase, ks, t);
      stage_B<K>(Wt, Bs, n0, NP, ks, t);
      wait_g2l();
      __syncthreads();
      slice_mma(As, Bs, wr, wc, lh, ll, a00, a01, a10, a11);
    }
    const int r0 = rowbase + wr * 32 + lh * 8;
    const int c0 = n0 + wc * 32 + ll;
#pragma unroll
    for (int j = 0; j < 8; ++j) {
      P[(size_t)(r0 + j) * NP + c0]           = (bf16_t)a00[j];
      P[(size_t)(r0 + j) * NP + c0 + 16]      = (bf16_t)a01[j];
      P[(size_t)(r0 + 16 + j) * NP + c0]      = (bf16_t)a10[j];
      P[(size_t)(r0 + 16 + j) * NP + c0 + 16] = (bf16_t)a11[j];
    }
  }
}

// -------------------- fused tier GEMM + online logsumexp -------------------
template<int K, int TIER>
__global__ __launch_bounds__(THREADS)
void tier_lse(const bf16_t* __restrict__ A,    // [8192][K]   K-fastest
              const bf16_t* __restrict__ Wt,   // [N][K]      K-fastest
              int N,
              const int* __restrict__ targets,
              const int* __restrict__ t2t, const int* __restrict__ t2i,
              float* __restrict__ row_m, float* __restrict__ row_s,
              float* __restrict__ row_t) {
  __shared__ bf16_t As[M_BLK * LDA];
  __shared__ bf16_t Bs[N_CHUNK * LDA];
  __shared__ float  pm[M_BLK][4], ps[M_BLK][4];     // per-wave-col partials
  __shared__ float  rm[M_BLK], rs[M_BLK], rt[M_BLK];
  __shared__ int    tcol[M_BLK];                    // target col or -1

  const int t = threadIdx.x;
  const int rowbase = blockIdx.x * M_BLK;
  const int wave = t >> 5, lane = t & 31;
  const int wr = wave >> 2, wc = wave & 3;
  const int lh = lane >> 4, ll = lane & 15;

  if (t < M_BLK) {
    int tg = targets[rowbase + t];
    tcol[t] = (t2t[tg] == TIER) ? t2i[tg] : -1;
    rm[t] = -1e30f; rs[t] = 0.f; rt[t] = 0.f;
  }

  const int nchunks = (N + N_CHUNK - 1) / N_CHUNK;
  for (int nc = 0; nc < nchunks; ++nc) {
    const int n0 = nc * N_CHUNK;
    v8f a00 = {}, a01 = {}, a10 = {}, a11 = {};
    for (int ks = 0; ks < K; ks += KS) {
      __syncthreads();
      if (K > KS || nc == 0) stage_A<K>(A, As, rowbase, ks, t);
      stage_B<K>(Wt, Bs, n0, N, ks, t);
      wait_g2l();
      __syncthreads();
      slice_mma(As, Bs, wr, wc, lh, ll, a00, a01, a10, a11);
    }
    // prefetch next chunk's weight rows
    if (nc + 1 < nchunks) {
      int n = n0 + N_CHUNK + (t >> 1);
      if (n < N) __builtin_prefetch(&Wt[(size_t)n * K + (t & 1) * 64], 0, 1);
    }
    // ---- per-row (max, sumexp) straight out of the accumulators ----
    const int cA = n0 + wc * 32 + ll;   // column of a*0 values
    const int cB = cA + 16;             // column of a*1 values
#pragma unroll
    for (int i = 0; i < 2; ++i) {
      v8f vA = i ? a10 : a00;
      v8f vB = i ? a11 : a01;
#pragma unroll
      for (int j = 0; j < 8; ++j) {
        const int row = wr * 32 + i * 16 + lh * 8 + j;
        float va = (cA < N) ? vA[j] : -1e30f;
        float vb = (cB < N) ? vB[j] : -1e30f;
        int tc = tcol[row];
        if (tc == cA) rt[row] = va;
        if (tc == cB) rt[row] = vb;
        float mx = fmaxf(va, vb);
        mx = fmaxf(mx, __shfl_xor(mx, 1));
        mx = fmaxf(mx, __shfl_xor(mx, 2));
        mx = fmaxf(mx, __shfl_xor(mx, 4));
        mx = fmaxf(mx, __shfl_xor(mx, 8));
        float se = __expf(va - mx) + __expf(vb - mx);
        se += __shfl_xor(se, 1);
        se += __shfl_xor(se, 2);
        se += __shfl_xor(se, 4);
        se += __shfl_xor(se, 8);
        if (ll == 0) { pm[row][wc] = mx; ps[row][wc] = se; }
      }
    }
    __syncthreads();
    // merge the 4 wave-column partials into the running row state
    if (t < M_BLK) {
      float m_c = pm[t][0], s_c = ps[t][0];
#pragma unroll
      for (int w = 1; w < 4; ++w) {
        float m2 = pm[t][w], s2 = ps[t][w];
        float m = fmaxf(m_c, m2);
        s_c = s_c * __expf(m_c - m) + s2 * __expf(m2 - m);
        m_c = m;
      }
      float m_old = rm[t];
      float m_new = fmaxf(m_old, m_c);
      rs[t] = rs[t] * __expf(m_old - m_new) + s_c * __expf(m_c - m_new);
      rm[t] = m_new;
    }
  }
  __syncthreads();

  // merge block state into global per-row state (tier kernels serialize)
  if (t < M_BLK) {
    const int g = rowbase + t;
    float gm = row_m[g], gs = row_s[g];
    float m_new = fmaxf(gm, rm[t]);
    row_s[g] = gs * __expf(gm - m_new) + rs[t] * __expf(rm[t] - m_new);
    row_m[g] = m_new;
    if (tcol[t] >= 0) row_t[g] = rt[t];
  }
}

// ------------------------------ final loss ---------------------------------
__global__ __launch_bounds__(THREADS)
void final_loss(const float* __restrict__ row_m, const float* __restrict__ row_s,
                const float* __restrict__ row_t, float* __restrict__ out, int n) {
  __shared__ float red[THREADS];
  float acc = 0.f;
  for (int i = threadIdx.x; i < n; i += THREADS)
    acc += row_m[i] + logf(row_s[i]) - row_t[i];
  red[threadIdx.x] = acc;
  __syncthreads();
  for (int s = THREADS / 2; s > 0; s >>= 1) {
    if (threadIdx.x < s) red[threadIdx.x] += red[threadIdx.x + s];
    __syncthreads();
  }
  if (threadIdx.x == 0) out[0] = red[0] / (float)n;
}

// ------------------------------- launcher ----------------------------------
extern "C" void kernel_launch(void* const* d_in, const int* in_sizes, int n_in,
                              void* d_out, int out_size, void* d_ws, size_t ws_size,
                              hipStream_t stream) {
  constexpr int BT = 8192, D = 1024;
  constexpr int N0 = 8192, N1 = 16384, N2 = 25681;
  constexpr int P1 = 256, P2 = 128;

  const float* h       = (const float*)d_in[0];
  const int*   targets = (const int*)d_in[1];   // int64 demoted to int32 by JAX default
  const int*   t2t     = (const int*)d_in[2];
  const int*   t2i     = (const int*)d_in[3];
  const float* W_head0 = (const float*)d_in[4];
  const float* W_proj1 = (const float*)d_in[5];
  const float* W_head1 = (const float*)d_in[6];
  const float* W_proj2 = (const float*)d_in[7];
  const float* W_head2 = (const float*)d_in[8];

  char* ws = (char*)d_ws;
  size_t off = 0;
  auto carve = [&](size_t bytes) { char* p = ws + off; off += (bytes + 255) & ~size_t(255); return p; };
  bf16_t* h_bf   = (bf16_t*)carve((size_t)BT * D * 2);   // [8192][1024]
  bf16_t* wt0    = (bf16_t*)carve((size_t)N0 * D * 2);   // [8192][1024]
  bf16_t* wtp1   = (bf16_t*)carve((size_t)P1 * D * 2);   // [256][1024]
  bf16_t* wt1    = (bf16_t*)carve((size_t)N1 * P1 * 2);  // [16384][256]
  bf16_t* wtp2   = (bf16_t*)carve((size_t)P2 * D * 2);   // [128][1024]
  bf16_t* wt2    = (bf16_t*)carve((size_t)N2 * P2 * 2);  // [25681][128]
  bf16_t* p1     = (bf16_t*)carve((size_t)BT * P1 * 2);
  bf16_t* p2     = (bf16_t*)carve((size_t)BT * P2 * 2);
  float*  row_m  = (float*)carve((size_t)BT * 4);
  float*  row_s  = (float*)carve((size_t)BT * 4);
  float*  row_t  = (float*)carve((size_t)BT * 4);

  // straight convert h; transpose-convert every weight matrix to K-fastest
  cvt_f32_bf16<<<(BT * D) / (THREADS * 8), THREADS, 0, stream>>>(h, h_bf, BT * D);
  auto cvtT = [&](const float* src, bf16_t* dst, int Kd, int Nd) {
    cvt_transpose<<<dim3((Nd + 63) / 64, Kd / 64), THREADS, 0, stream>>>(src, dst, Kd, Nd);
  };
  cvtT(W_head0, wt0,  D,  N0);
  cvtT(W_proj1, wtp1, D,  P1);
  cvtT(W_head1, wt1,  P1, N1);
  cvtT(W_proj2, wtp2, D,  P2);
  cvtT(W_head2, wt2,  P2, N2);

  init_state<<<BT / THREADS, THREADS, 0, stream>>>(row_m, row_s, row_t, BT);

  proj_gemm<P1><<<BT / M_BLK, THREADS, 0, stream>>>(h_bf, wtp1, p1);
  proj_gemm<P2><<<BT / M_BLK, THREADS, 0, stream>>>(h_bf, wtp2, p2);

  tier_lse<1024, 0><<<BT / M_BLK, THREADS, 0, stream>>>(h_bf, wt0, N0, targets, t2t, t2i, row_m, row_s, row_t);
  tier_lse<256,  1><<<BT / M_BLK, THREADS, 0, stream>>>(p1,   wt1, N1, targets, t2t, t2i, row_m, row_s, row_t);
  tier_lse<128,  2><<<BT / M_BLK, THREADS, 0, stream>>>(p2,   wt2, N2, targets, t2t, t2i, row_m, row_s, row_t);

  final_loss<<<1, THREADS, 0, stream>>>(row_m, row_s, row_t, (float*)d_out, BT);
}
